// BertSelfAttention_71871982731335
// MI455X (gfx1250) — matrix operand order repaired and checked by reading, not compiled, not run on hardware
//
#include <hip/hip_runtime.h>
#include <hip/hip_bf16.h>
#include <math.h>

typedef unsigned short u16;
typedef unsigned int   u32;
typedef __attribute__((ext_vector_type(16))) __bf16 v16bf;
typedef __attribute__((ext_vector_type(8)))  float  v8f;

#define Bn   4
#define Sn   2048
#define Hn   1024
#define NH   16
#define HD   64
// total logits scale: q * hd^-0.5 pre-rotary (linear -> commutes) * 1/sqrt(hd) post => 1/64
#define LOGIT_SCALE 0.015625f

union Frag { u32 u[8]; v16bf v; };

__device__ __forceinline__ float bf2f(u16 h) {
    union { u32 i; float f; } x; x.i = ((u32)h) << 16; return x.f;
}
__device__ __forceinline__ u16 f2bf(float f) {
    union { float f; u32 i; } x; x.f = f;
    u32 r = (x.i + 0x7fffu + ((x.i >> 16) & 1u)) >> 16;   // RNE
    return (u16)r;
}

// ---- CDNA5 async global->LDS copy (ASYNCcnt path, bypasses VGPRs) ----------
// per-lane: each lane copies from its global address to its LDS address, 16B per lane per instruction.
__device__ __forceinline__ void async_b128(const void* g, void* l) {
    asm volatile("global_load_async_to_lds_b128 %0, %1, off"
                 :: "v"((u32)(uintptr_t)l),                 // LDS byte offset
                    "v"((unsigned long long)(uintptr_t)g)   // 64-bit global addr
                 : "memory");
}
__device__ __forceinline__ void wait_async0() {
    asm volatile("s_wait_asynccnt 0x0" ::: "memory");
}
__device__ __forceinline__ void wait_async4() {
    asm volatile("s_wait_asynccnt 0x4" ::: "memory");
}
__device__ __forceinline__ void wait_async8() {
    asm volatile("s_wait_asynccnt 0x8" ::: "memory");
}

// ---------------------------------------------------------------------------
// 1a) fp32 -> bf16 conversion (plain)
// ---------------------------------------------------------------------------
__global__ void f32_to_bf16_kernel(const float* __restrict__ src,
                                   u16* __restrict__ dst, int n) {
    int i = blockIdx.x * 256 + threadIdx.x;
    if (i < n) dst[i] = f2bf(src[i]);
}

// 1b) fp32 -> bf16 with transpose: dst[n][k] = src[k][n]  (1024x1024)
__global__ void f32_to_bf16_T_kernel(const float* __restrict__ src,
                                     u16* __restrict__ dst) {
    int i = blockIdx.x * 256 + threadIdx.x;   // < 1048576
    int k = i & 1023, n = i >> 10;
    dst[(size_t)n * Hn + k] = f2bf(src[(size_t)k * Hn + n]);
}

// ---------------------------------------------------------------------------
// 2) Y = X @ W + bias; X bf16 [B*S,H], Wt bf16 pre-transposed [n][k].
//    Output scattered to head layout; vtrans=1 stores [B,nh,hd,S] (for V).
//    block = 256 threads (8 waves), block tile 128x128, wave tile 64x32.
//    Both operands staged to LDS with double-buffered async B128 copies.
// ---------------------------------------------------------------------------
__global__ __launch_bounds__(256) void gemm_bias_heads_kernel(
    const u16* __restrict__ X, const u16* __restrict__ Wt,
    const float* __restrict__ bias, u16* __restrict__ out, int vtrans)
{
    __shared__ u16 Xs[2][128 * 32];  // [row][k]
    __shared__ u16 Ws[2][128 * 32];  // [n][k]

    const int tid  = threadIdx.x;
    const int wave = tid >> 5, lane = tid & 31;
    const int l16  = lane & 15, lh = lane >> 4;
    const int mBase = blockIdx.x * 128;
    const int nBase = blockIdx.y * 128;
    const int mw = (wave >> 2) * 64;
    const int nw = (wave & 3) * 32;

    v8f acc[4][2];
#pragma unroll
    for (int i = 0; i < 4; ++i)
#pragma unroll
        for (int j = 0; j < 2; ++j)
#pragma unroll
            for (int v = 0; v < 8; ++v) acc[i][j][v] = 0.0f;

    // issue one 128x32 tile pair into LDS buffer `buf` (4 async ops/thread)
    auto stage = [&](int k0, int buf) {
#pragma unroll
        for (int j = 0; j < 2; ++j) {
            int c = tid + j * 256, r = c >> 2, seg = c & 3;
            async_b128(X + (size_t)(mBase + r) * Hn + k0 + seg * 8,
                       &Xs[buf][r * 32 + seg * 8]);
        }
#pragma unroll
        for (int j = 0; j < 2; ++j) {
            int c = tid + j * 256, r = c >> 2, seg = c & 3;
            async_b128(Wt + (size_t)(nBase + r) * Hn + k0 + seg * 8,
                       &Ws[buf][r * 32 + seg * 8]);
        }
    };

    stage(0, 0);
    for (int it = 0; it < Hn / 32; ++it) {
        const int buf = it & 1;
        if (it + 1 < Hn / 32) {
            stage((it + 1) * 32, buf ^ 1);  // prefetch next tile
            wait_async4();                  // wait out current tile only
        } else {
            wait_async0();
        }
        __syncthreads();

        Frag a[4];
#pragma unroll
        for (int sm = 0; sm < 4; ++sm) {
            const u32* rp = (const u32*)(&Xs[buf][(mw + sm * 16 + l16) * 32]);
#pragma unroll
            for (int v = 0; v < 8; ++v) {
                int kk = lh * 16 + 2 * v;
                a[sm].u[v] = rp[kk >> 1];
            }
        }
        Frag b[2];
#pragma unroll
        for (int sn = 0; sn < 2; ++sn) {
            const u32* np = (const u32*)(&Ws[buf][(nw + sn * 16 + l16) * 32]);
#pragma unroll
            for (int v = 0; v < 8; ++v) b[sn].u[v] = np[lh * 8 + v];
        }
#pragma unroll
        for (int sm = 0; sm < 4; ++sm)
#pragma unroll
            for (int sn = 0; sn < 2; ++sn)
                acc[sm][sn] = __builtin_amdgcn_wmma_f32_16x16x32_bf16(
                    false, a[sm].v, false, b[sn].v, (short)0, acc[sm][sn],
                    false, false);
        __syncthreads();
    }

    // ---- epilogue: bias + head-layout scatter ----
#pragma unroll
    for (int sn = 0; sn < 2; ++sn) {
        int col = nBase + nw + sn * 16 + l16;
        float bv = bias[col];
        int h = col >> 6, d = col & 63;
#pragma unroll
        for (int sm = 0; sm < 4; ++sm) {
#pragma unroll
            for (int v = 0; v < 8; ++v) {
                int row = mBase + mw + sm * 16 + v + 8 * lh;  // bs index
                int bb = row >> 11, s = row & 2047;
                u16 val = f2bf(acc[sm][sn][v] + bv);
                if (!vtrans)
                    out[(((size_t)bb * NH + h) * Sn + s) * HD + d] = val;
                else  // V stored transposed: [B, nh, hd, S]
                    out[(((size_t)bb * NH + h) * HD + d) * Sn + s] = val;
            }
        }
    }
}

// ---------------------------------------------------------------------------
// 3) in-place rotary on a [B, nh, S, hd] bf16 buffer; thread per (bh,s,d<32)
// ---------------------------------------------------------------------------
__global__ void rotary_kernel(u16* __restrict__ p) {
    int i = blockIdx.x * 256 + threadIdx.x;     // < B*NH*S*32
    int d  = i & 31;
    int s  = (i >> 5) & 2047;
    int bh = i >> 16;
    size_t base = ((size_t)bh * Sn + s) * HD;
    float inv = __expf(-(float)d * (9.210340371976184f / 32.0f)); // 1e4^{-d/32}
    float ang = (float)s * inv;
    float c = __cosf(ang), sn = __sinf(ang);
    float x1 = bf2f(p[base + d]);
    float x2 = bf2f(p[base + d + 32]);
    p[base + d]      = f2bf(x1 * c - x2 * sn);
    p[base + d + 32] = f2bf(x2 * c + x1 * sn);
}

// ---------------------------------------------------------------------------
// 4) flash attention: grid (S/64, B*nh), 128 threads = 4 waves, each wave
//    owns 16 query rows. K [bh][s][d]; V pre-transposed [bh][d][s] so both
//    tiles stage via double-buffered async B128 copies.
// ---------------------------------------------------------------------------
__global__ __launch_bounds__(128) void flash_attn_kernel(
    const u16* __restrict__ Q, const u16* __restrict__ K,
    const u16* __restrict__ Vt, float* __restrict__ out)
{
    __shared__ u16 Ks[2][64 * 64];   // [key][d]
    __shared__ u16 Vs[2][64 * 64];   // [d][key]
    __shared__ u16 Ps[4][16 * 64];   // per-wave P tile

    const int tid = threadIdx.x, wave = tid >> 5, lane = tid & 31;
    const int l16 = lane & 15, lh = lane >> 4;
    const int bh = blockIdx.y;
    const int bb = bh >> 4, h = bh & 15;
    const int q0 = blockIdx.x * 64;
    const u16* Qh = Q  + (size_t)bh * Sn * HD;
    const u16* Kh = K  + (size_t)bh * Sn * HD;
    const u16* Vh = Vt + (size_t)bh * HD * Sn;

    // Q A-fragments (16 rows x 64 d), two K=32 steps, loaded once
    Frag qa[2];
    {
        const u32* qp = (const u32*)(Qh + (size_t)(q0 + wave * 16 + l16) * HD);
#pragma unroll
        for (int s = 0; s < 2; ++s)
#pragma unroll
            for (int v = 0; v < 8; ++v) {
                int kk = s * 32 + lh * 16 + 2 * v;
                qa[s].u[v] = qp[kk >> 1];
            }
    }

    float m_[8], l_[8];
    v8f o[4];
#pragma unroll
    for (int v = 0; v < 8; ++v) { m_[v] = -1e30f; l_[v] = 0.0f; }
#pragma unroll
    for (int t = 0; t < 4; ++t)
#pragma unroll
        for (int v = 0; v < 8; ++v) o[t][v] = 0.0f;

    // issue one K+V tile pair into buffer `buf` (8 async ops/thread)
    auto stage = [&](int key0, int buf) {
#pragma unroll
        for (int j = 0; j < 4; ++j) {
            int c = tid + j * 128, row = c >> 3, seg = c & 7;
            async_b128(Kh + (size_t)(key0 + row) * HD + seg * 8,
                       &Ks[buf][row * 64 + seg * 8]);
        }
#pragma unroll
        for (int j = 0; j < 4; ++j) {
            int c = tid + j * 128, row = c >> 3, seg = c & 7;  // row = d
            async_b128(Vh + (size_t)row * Sn + key0 + seg * 8,
                       &Vs[buf][row * 64 + seg * 8]);
        }
    };

    stage(0, 0);
    for (int kt = 0; kt < Sn / 64; ++kt) {
        const int buf = kt & 1;
        if (kt + 1 < Sn / 64) {
            stage((kt + 1) * 64, buf ^ 1);  // prefetch next tile pair
            wait_async8();                  // retire current tile's 8 ops
        } else {
            wait_async0();
        }
        __syncthreads();

        // ---- logits: S = Q K^T ----
        v8f sacc[4];
#pragma unroll
        for (int t = 0; t < 4; ++t) {
            Frag kb0, kb1;
            const u32* kp = (const u32*)(&Ks[buf][(t * 16 + l16) * 64]);
#pragma unroll
            for (int v = 0; v < 8; ++v) {
                kb0.u[v] = kp[lh * 8 + v];        // d = 0..31
                kb1.u[v] = kp[16 + lh * 8 + v];   // d = 32..63
            }
            v8f z;
#pragma unroll
            for (int v = 0; v < 8; ++v) z[v] = 0.0f;
            sacc[t] = __builtin_amdgcn_wmma_f32_16x16x32_bf16(
                false, qa[0].v, false, kb0.v, (short)0, z, false, false);
            sacc[t] = __builtin_amdgcn_wmma_f32_16x16x32_bf16(
                false, qa[1].v, false, kb1.v, (short)0, sacc[t], false, false);
        }

        // ---- online softmax (rows live in 16-lane halves) ----
        float mnew[8], alpha[8], rsum[8];
#pragma unroll
        for (int v = 0; v < 8; ++v) {
            float x = fmaxf(fmaxf(sacc[0][v], sacc[1][v]),
                            fmaxf(sacc[2][v], sacc[3][v])) * LOGIT_SCALE;
            x = fmaxf(x, __shfl_xor(x, 1, 32));
            x = fmaxf(x, __shfl_xor(x, 2, 32));
            x = fmaxf(x, __shfl_xor(x, 4, 32));
            x = fmaxf(x, __shfl_xor(x, 8, 32));
            mnew[v]  = fmaxf(m_[v], x);
            alpha[v] = __expf(m_[v] - mnew[v]);
            rsum[v]  = 0.0f;
        }
        u16* Pw = Ps[wave];
#pragma unroll
        for (int t = 0; t < 4; ++t)
#pragma unroll
            for (int v = 0; v < 8; ++v) {
                float pv = __expf(sacc[t][v] * LOGIT_SCALE - mnew[v]);
                rsum[v] += pv;
                Pw[(v + 8 * lh) * 64 + t * 16 + l16] = f2bf(pv);
            }
#pragma unroll
        for (int v = 0; v < 8; ++v) {
            float r = rsum[v];
            r += __shfl_xor(r, 1, 32);
            r += __shfl_xor(r, 2, 32);
            r += __shfl_xor(r, 4, 32);
            r += __shfl_xor(r, 8, 32);
            l_[v] = l_[v] * alpha[v] + r;
            m_[v] = mnew[v];
        }
#pragma unroll
        for (int t = 0; t < 4; ++t)
#pragma unroll
            for (int v = 0; v < 8; ++v) o[t][v] *= alpha[v];

        // ---- O += P @ V ----
        Frag pa[2];
        {
            const u32* pp = (const u32*)(Pw + l16 * 64);
#pragma unroll
            for (int s = 0; s < 2; ++s)
#pragma unroll
                for (int v = 0; v < 8; ++v) {
                    int kk = s * 32 + lh * 16 + 2 * v;
                    pa[s].u[v] = pp[kk >> 1];
                }
        }
#pragma unroll
        for (int t = 0; t < 4; ++t) {
            Frag vb0, vb1;
            const u32* vp = (const u32*)(&Vs[buf][(t * 16 + l16) * 64]);
#pragma unroll
            for (int v = 0; v < 8; ++v) {
                vb0.u[v] = vp[lh * 8 + v];        // keys 0..31
                vb1.u[v] = vp[16 + lh * 8 + v];   // keys 32..63
            }
            o[t] = __builtin_amdgcn_wmma_f32_16x16x32_bf16(
                false, pa[0].v, false, vb0.v, (short)0, o[t], false, false);
            o[t] = __builtin_amdgcn_wmma_f32_16x16x32_bf16(
                false, pa[1].v, false, vb1.v, (short)0, o[t], false, false);
        }
        __syncthreads();
    }

    // ---- normalize + store fp32 to [B,S,H] ----
#pragma unroll
    for (int t = 0; t < 4; ++t) {
#pragma unroll
        for (int v = 0; v < 8; ++v) {
            int srow = q0 + wave * 16 + v + 8 * lh;
            int d = t * 16 + l16;
            out[((size_t)bb * Sn + srow) * Hn + h * HD + d] = o[t][v] / l_[v];
        }
    }
}

// ---------------------------------------------------------------------------
// launcher
// ---------------------------------------------------------------------------
extern "C" void kernel_launch(void* const* d_in, const int* in_sizes, int n_in,
                              void* d_out, int out_size, void* d_ws,
                              size_t ws_size, hipStream_t stream) {
    const float* hs = (const float*)d_in[0];
    const float* Wq = (const float*)d_in[1];
    const float* bq = (const float*)d_in[2];
    const float* Wk = (const float*)d_in[3];
    const float* bk = (const float*)d_in[4];
    const float* Wv = (const float*)d_in[5];
    const float* bv = (const float*)d_in[6];
    float* out = (float*)d_out;

    const size_t nX = (size_t)Bn * Sn * Hn;   // 8,388,608
    const size_t nW = (size_t)Hn * Hn;        // 1,048,576

    char* ws = (char*)d_ws;
    u16* Xb  = (u16*)(ws);
    u16* Wqb = (u16*)(ws + nX * 2);               // transposed [n][k]
    u16* Wkb = (u16*)(ws + nX * 2 + nW * 2);
    u16* Wvb = (u16*)(ws + nX * 2 + nW * 4);
    u16* Qb  = (u16*)(ws + nX * 2 + nW * 6);      // [B,nh,S,hd]
    u16* Kb  = (u16*)(ws + nX * 4 + nW * 6);      // [B,nh,S,hd]
    u16* Vb  = (u16*)(ws + nX * 6 + nW * 6);      // [B,nh,hd,S] (transposed)

    // 1) fp32 -> bf16 (weights transposed for contiguous B-operand staging)
    f32_to_bf16_kernel<<<(int)(nX / 256), 256, 0, stream>>>(hs, Xb, (int)nX);
    f32_to_bf16_T_kernel<<<(int)(nW / 256), 256, 0, stream>>>(Wq, Wqb);
    f32_to_bf16_T_kernel<<<(int)(nW / 256), 256, 0, stream>>>(Wk, Wkb);
    f32_to_bf16_T_kernel<<<(int)(nW / 256), 256, 0, stream>>>(Wv, Wvb);

    // 2) QKV projections (WMMA GEMM, async double-buffered staging)
    dim3 ggrid(Bn * Sn / 128, Hn / 128);
    gemm_bias_heads_kernel<<<ggrid, 256, 0, stream>>>(Xb, Wqb, bq, Qb, 0);
    gemm_bias_heads_kernel<<<ggrid, 256, 0, stream>>>(Xb, Wkb, bk, Kb, 0);
    gemm_bias_heads_kernel<<<ggrid, 256, 0, stream>>>(Xb, Wvb, bv, Vb, 1);

    // 3) rotary in place on Q and K
    int rthreads = Bn * NH * Sn * 32;  // 4,194,304
    rotary_kernel<<<rthreads / 256, 256, 0, stream>>>(Qb);
    rotary_kernel<<<rthreads / 256, 256, 0, stream>>>(Kb);

    // 4) flash attention (async double-buffered K/V tiles)
    flash_attn_kernel<<<dim3(Sn / 64, Bn * NH), 128, 0, stream>>>(Qb, Kb, Vb,
                                                                  out);
}
